// DecoderCell_3573412790665
// MI455X (gfx1250) — matrix operand order repaired and checked by reading, not compiled, not run on hardware
//
#include <hip/hip_runtime.h>

// ---------------- CDNA5 WMMA types ----------------
typedef __attribute__((ext_vector_type(16))) __bf16 v16bf;
typedef __attribute__((ext_vector_type(8)))  float  v8f;

union BF16Frag {
  v16bf  v;
  __bf16 h[16];
  uint4  q[2];
};

__device__ __forceinline__ v8f wmma_bf16(v16bf a, v16bf b, v8f c) {
  return __builtin_amdgcn_wmma_f32_16x16x32_bf16(
      false, a, false, b, (short)0, c, false, false);
}

__device__ __forceinline__ float sigm(float v) { return 1.0f / (1.0f + __expf(-v)); }

// ---------------- problem constants ----------------
#define GEN_DIM   200
#define CON_DIM   128
#define CO_DIM    4
#define LATENT    64
#define X_LD      272   // 2*CI_ENC + EXT
#define H_LD      420   // STATE_DIM
#define GEN_IN    20
#define CON_IN    320
#define CLIPV     5.0f
#define GPAD      208   // gen gate rows padded 200 -> 208 (16-multiple)
#define KPAD      224   // gen/fac K padded 200 -> 224 (32-multiple)

// ---- bf16 weight workspace layout (element offsets into __bf16* ws) ----
#define WOFF_CONWIH  0                          // [384][320]
#define WOFF_CONWHH  (WOFF_CONWIH + 384*320)    // [384][128]
#define WOFF_GENWIH  (WOFF_CONWHH + 384*128)    // [3*208][32]
#define WOFF_GENWHH  (WOFF_GENWIH + 3*GPAD*32)  // [3*208][224]
#define WOFF_COW     (WOFF_GENWHH + 3*GPAD*KPAD)// [16][128]
#define WOFF_FACW    (WOFF_COW + 16*128)        // [64][224] (pre-normalized)
#define WELEMS       (WOFF_FACW + LATENT*KPAD)
#define WS_INVN_BYTES (WELEMS * 2)              // 64 floats of inv-norms after bf16 area

// ---------------- fragment loaders ----------------
// A fragment (16x32 bf16) from row-major fp32 with K guard.
__device__ __forceinline__ v16bf loadA_g(const float* __restrict__ X, int ldx,
                                         int kfrag, int kMax) {
  int lane = threadIdx.x & 31;
  int m    = lane & 15;
  int h8   = (lane >> 4) << 3;
  const float* row = X + (size_t)m * ldx;
  BF16Frag a;
#pragma unroll
  for (int c = 0; c < 2; ++c) {
    int k0 = kfrag * 32 + c * 16 + h8;
    if (k0 + 8 <= kMax) {
      float4 f0 = *(const float4*)(row + k0);
      float4 f1 = *(const float4*)(row + k0 + 4);
      a.h[c*8+0]=(__bf16)f0.x; a.h[c*8+1]=(__bf16)f0.y;
      a.h[c*8+2]=(__bf16)f0.z; a.h[c*8+3]=(__bf16)f0.w;
      a.h[c*8+4]=(__bf16)f1.x; a.h[c*8+5]=(__bf16)f1.y;
      a.h[c*8+6]=(__bf16)f1.z; a.h[c*8+7]=(__bf16)f1.w;
    } else {
#pragma unroll
      for (int e = 0; e < 8; ++e) {
        float v = (k0 + e < kMax) ? row[k0 + e] : 0.0f;
        a.h[c*8+e] = (__bf16)v;
      }
    }
  }
  return a.v;
}

// A fragment from per-wave LDS bf16 staging (padded, unguarded).
__device__ __forceinline__ v16bf loadA_lds(const __bf16* S, int ld, int kfrag) {
  int lane = threadIdx.x & 31;
  int m    = lane & 15;
  int h8   = (lane >> 4) << 3;
  const __bf16* p = S + m * ld + kfrag * 32;
  BF16Frag a;
  a.q[0] = *(const uint4*)(p + h8);
  a.q[1] = *(const uint4*)(p + 16 + h8);
  return a.v;
}

// B fragment (32x16 bf16) from pre-converted, pre-padded bf16 weights:
// two b128 loads per lane, no conversion, no guards.
__device__ __forceinline__ v16bf loadB_bf(const __bf16* __restrict__ W, int ldw,
                                          int row0, int kfrag) {
  int lane = threadIdx.x & 31;
  int n    = lane & 15;
  int k0   = kfrag * 32 + ((lane >> 4) << 3);
  const __bf16* p = W + (size_t)(row0 + n) * ldw + k0;
  BF16Frag b;
  b.q[0] = *(const uint4*)(p);
  b.q[1] = *(const uint4*)(p + 16);
  return b.v;
}

// C/D-layout gather from global fp32.
__device__ __forceinline__ v8f loadC(const float* __restrict__ X, int ldx,
                                     int col0, int colMax) {
  int lane = threadIdx.x & 31;
  int n    = lane & 15;
  int mo   = (lane >> 4) << 3;
  int col  = col0 + n;
  v8f c = {};
  if (col < colMax) {
#pragma unroll
    for (int i = 0; i < 8; ++i) c[i] = X[(size_t)(mo + i) * ldx + col];
  }
  return c;
}

// ---------------- prep kernels (run once, tiny) ----------------
__global__ void prep_norms(const float* __restrict__ facW, float* __restrict__ invn) {
  int r = threadIdx.x;
  if (r < LATENT) {
    const float* p = facW + r * GEN_DIM;
    float s = 0.0f;
    for (int k = 0; k < GEN_DIM; ++k) { float v = p[k]; s += v * v; }
    invn[r] = 1.0f / fmaxf(sqrtf(s), 1e-12f);
  }
}

__global__ void prep_weights(const float* __restrict__ conWih,
                             const float* __restrict__ conWhh,
                             const float* __restrict__ genWih,
                             const float* __restrict__ genWhh,
                             const float* __restrict__ coW,
                             const float* __restrict__ facW,
                             const float* __restrict__ invn,
                             __bf16* __restrict__ ws) {
  int gid = blockIdx.x * blockDim.x + threadIdx.x;
  int str = gridDim.x * blockDim.x;
  for (int i = gid; i < 384 * CON_IN; i += str)          // con_w_ih (dense)
    ws[WOFF_CONWIH + i] = (__bf16)conWih[i];
  for (int i = gid; i < 384 * CON_DIM; i += str)         // con_w_hh (dense)
    ws[WOFF_CONWHH + i] = (__bf16)conWhh[i];
  for (int i = gid; i < 3 * GPAD * 32; i += str) {       // gen_w_ih padded rows+K
    int r = i >> 5, c = i & 31;
    int g = r / GPAD, rl = r % GPAD;
    float v = (rl < GEN_DIM && c < GEN_IN) ? genWih[(g * GEN_DIM + rl) * GEN_IN + c] : 0.0f;
    ws[WOFF_GENWIH + i] = (__bf16)v;
  }
  for (int i = gid; i < 3 * GPAD * KPAD; i += str) {     // gen_w_hh padded rows+K
    int r = i / KPAD, c = i % KPAD;
    int g = r / GPAD, rl = r % GPAD;
    float v = (rl < GEN_DIM && c < GEN_DIM) ? genWhh[(g * GEN_DIM + rl) * GEN_DIM + c] : 0.0f;
    ws[WOFF_GENWHH + i] = (__bf16)v;
  }
  for (int i = gid; i < 16 * CON_DIM; i += str) {        // co_w padded rows 8->16
    int r = i >> 7, c = i & 127;
    float v = (r < 2 * CO_DIM) ? coW[r * CON_DIM + c] : 0.0f;
    ws[WOFF_COW + i] = (__bf16)v;
  }
  for (int i = gid; i < LATENT * KPAD; i += str) {       // fac_w normalized + K pad
    int r = i / KPAD, c = i % KPAD;
    float v = (c < GEN_DIM) ? facW[r * GEN_DIM + c] * invn[r] : 0.0f;
    ws[WOFF_FACW + i] = (__bf16)v;
  }
}

// ---------------- main kernel ----------------
__global__ __launch_bounds__(128) void decoder_cell(
    const float* __restrict__ x,   const float* __restrict__ h0,
    const float* __restrict__ eps,
    const float* __restrict__ genBih, const float* __restrict__ genBhh,
    const float* __restrict__ conBih, const float* __restrict__ conBhh,
    const float* __restrict__ coB,
    const __bf16* __restrict__ wbf,
    float* __restrict__ out)
{
  __shared__ __align__(16) __bf16 s_conS[4][16 * CON_DIM];   // new con_state
  __shared__ __align__(16) __bf16 s_geni[4][16 * 32];        // gen_input, K=32 pad
  __shared__ float  s_coS[4][16 * 8];                        // co mean/std
  __shared__ __align__(16) __bf16 s_genS[4][16 * KPAD];      // new gen_state, K pad

  const int tid  = threadIdx.x;
  const int wave = tid >> 5;
  const int lane = tid & 31;
  const int m0   = blockIdx.x * 64 + wave * 16;

  const __bf16* conWih = wbf + WOFF_CONWIH;
  const __bf16* conWhh = wbf + WOFF_CONWHH;
  const __bf16* genWih = wbf + WOFF_GENWIH;
  const __bf16* genWhh = wbf + WOFF_GENWHH;
  const __bf16* coWb   = wbf + WOFF_COW;
  const __bf16* facWb  = wbf + WOFF_FACW;

  __bf16* genS = s_genS[wave];
  for (int i = lane; i < 16 * KPAD; i += 32) genS[i] = (__bf16)0.0f;  // K-pad zeros

  const float* xrow = x  + (size_t)m0 * X_LD;
  const float* hrow = h0 + (size_t)m0 * H_LD;
  const int n  = lane & 15;
  const int mo = (lane >> 4) << 3;

  // ========== Phase 1: controller GRU (320 -> 128) ==========
  // r/z gates: input-side and hidden-side GEMMs share one accumulator each
  // (gi+gh is all that's ever used); only the n-gate needs i_n / h_n split.
  v16bf acn[10], ahc[4];
#pragma unroll
  for (int k = 0; k < 8; ++k) acn[k] = loadA_g(xrow, X_LD, k, 256);               // ci
#pragma unroll
  for (int k = 0; k < 2; ++k) acn[8 + k] = loadA_g(hrow + 356, H_LD, k, LATENT);  // factor
#pragma unroll
  for (int k = 0; k < 4; ++k) ahc[k] = loadA_g(hrow + 200, H_LD, k, CON_DIM);     // h_con

  __bf16* conS = s_conS[wave];
#pragma unroll 1
  for (int j = 0; j < 8; ++j) {
    v8f ar = {}, az = {}, ain = {}, ahn = {};
#pragma unroll
    for (int kk = 0; kk < 10; ++kk) {
      ar  = wmma_bf16(acn[kk], loadB_bf(conWih, CON_IN,       j*16, kk), ar);
      az  = wmma_bf16(acn[kk], loadB_bf(conWih, CON_IN, 128 + j*16, kk), az);
      ain = wmma_bf16(acn[kk], loadB_bf(conWih, CON_IN, 256 + j*16, kk), ain);
    }
#pragma unroll
    for (int kk = 0; kk < 4; ++kk) {
      ar  = wmma_bf16(ahc[kk], loadB_bf(conWhh, CON_DIM,       j*16, kk), ar);
      az  = wmma_bf16(ahc[kk], loadB_bf(conWhh, CON_DIM, 128 + j*16, kk), az);
      ahn = wmma_bf16(ahc[kk], loadB_bf(conWhh, CON_DIM, 256 + j*16, kk), ahn);
    }
    int col = j * 16 + n;
    float br = conBih[col]       + conBhh[col];
    float bz = conBih[128 + col] + conBhh[128 + col];
    float bin = conBih[256 + col], bhn = conBhh[256 + col];
    v8f hold = loadC(hrow + 200, H_LD, j * 16, CON_DIM);
#pragma unroll
    for (int i = 0; i < 8; ++i) {
      float r  = sigm(ar[i] + br);
      float z  = sigm(az[i] + bz);
      float nn = tanhf(ain[i] + bin + r * (ahn[i] + bhn));
      float hc = (1.0f - z) * nn + z * hold[i];
      hc = fminf(fmaxf(hc, -CLIPV), CLIPV);
      out[(size_t)(m0 + mo + i) * H_LD + 200 + col] = hc;
      conS[(mo + i) * CON_DIM + col] = (__bf16)hc;
    }
  }

  // ========== Phase 2: co head + reparameterized sample ==========
  {
    v8f co = {};
#pragma unroll
    for (int kk = 0; kk < 4; ++kk)
      co = wmma_bf16(loadA_lds(conS, CON_DIM, kk), loadB_bf(coWb, CON_DIM, 0, kk), co);
    float* coS = s_coS[wave];
    if (n < 8) {
      float bb = coB[n];
#pragma unroll
      for (int i = 0; i < 8; ++i) {
        float v = co[i] + bb;
        int m = mo + i;
        if (n < CO_DIM) {
          out[(size_t)(m0 + m) * H_LD + 328 + n] = v;      // co_mean
          coS[m * 8 + n] = v;
        } else {
          float sd = __expf(0.5f * v);                     // co_std
          out[(size_t)(m0 + m) * H_LD + 332 + (n - CO_DIM)] = sd;
          coS[m * 8 + n] = sd;
        }
      }
    }
    __bf16* geni = s_geni[wave];
    float* coSr = s_coS[wave];
    int c = lane;
    for (int m = 0; m < 16; ++m) {
      float v;
      if (c < CO_DIM)
        v = coSr[m * 8 + c] + coSr[m * 8 + CO_DIM + c] * eps[(size_t)(m0 + m) * CO_DIM + c];
      else if (c < GEN_IN)
        v = x[(size_t)(m0 + m) * X_LD + 256 + (c - CO_DIM)];
      else
        v = 0.0f;
      geni[m * 32 + c] = (__bf16)v;
      if (c < GEN_IN) out[(size_t)(m0 + m) * H_LD + 336 + c] = v;   // gen_input
    }
  }

  // ========== Phase 3: generator GRU (20 -> 200) ==========
  v16bf agin = loadA_lds(s_geni[wave], 32, 0);
  v16bf ahg[7];
#pragma unroll
  for (int k = 0; k < 7; ++k) ahg[k] = loadA_g(hrow, H_LD, k, GEN_DIM);

#pragma unroll 1
  for (int j = 0; j < 13; ++j) {
    v8f ar = {}, az = {}, ain = {}, ahn = {};
    ar  = wmma_bf16(agin, loadB_bf(genWih, 32,          j*16, 0), ar);
    az  = wmma_bf16(agin, loadB_bf(genWih, 32,   GPAD + j*16, 0), az);
    ain = wmma_bf16(agin, loadB_bf(genWih, 32, 2*GPAD + j*16, 0), ain);
#pragma unroll
    for (int kk = 0; kk < 7; ++kk) {
      ar  = wmma_bf16(ahg[kk], loadB_bf(genWhh, KPAD,          j*16, kk), ar);
      az  = wmma_bf16(ahg[kk], loadB_bf(genWhh, KPAD,   GPAD + j*16, kk), az);
      ahn = wmma_bf16(ahg[kk], loadB_bf(genWhh, KPAD, 2*GPAD + j*16, kk), ahn);
    }
    int col = j * 16 + n;
    bool valid = (col < GEN_DIM);
    float br = 0.f, bz = 0.f, bin = 0.f, bhn = 0.f;
    if (valid) {
      br  = genBih[col]       + genBhh[col];
      bz  = genBih[200 + col] + genBhh[200 + col];
      bin = genBih[400 + col];
      bhn = genBhh[400 + col];
    }
    v8f hold = loadC(hrow, H_LD, j * 16, GEN_DIM);
#pragma unroll
    for (int i = 0; i < 8; ++i) {
      float r  = sigm(ar[i] + br);
      float z  = sigm(az[i] + bz);
      float nn = tanhf(ain[i] + bin + r * (ahn[i] + bhn));
      float hc = (1.0f - z) * nn + z * hold[i];
      hc = fminf(fmaxf(hc, -CLIPV), CLIPV);
      if (valid) {
        out[(size_t)(m0 + mo + i) * H_LD + col] = hc;       // gen_state
        genS[(mo + i) * KPAD + col] = (__bf16)hc;
      }
    }
  }

  // ========== Phase 4: factor = gen_state @ normed(fac_w).T ==========
  v16bf agS[7];
#pragma unroll
  for (int k = 0; k < 7; ++k) agS[k] = loadA_lds(genS, KPAD, k);
#pragma unroll 1
  for (int j = 0; j < 4; ++j) {
    v8f fa = {};
#pragma unroll
    for (int kk = 0; kk < 7; ++kk)
      fa = wmma_bf16(agS[kk], loadB_bf(facWb, KPAD, j * 16, kk), fa);
#pragma unroll
    for (int i = 0; i < 8; ++i)
      out[(size_t)(m0 + mo + i) * H_LD + 356 + j * 16 + n] = fa[i]; // factor
  }
}

// ---------------- host entry ----------------
extern "C" void kernel_launch(void* const* d_in, const int* in_sizes, int n_in,
                              void* d_out, int out_size, void* d_ws, size_t ws_size,
                              hipStream_t stream) {
  const float* x        = (const float*)d_in[0];
  const float* h0       = (const float*)d_in[1];
  const float* eps      = (const float*)d_in[2];
  const float* gen_w_ih = (const float*)d_in[3];
  const float* gen_w_hh = (const float*)d_in[4];
  const float* gen_b_ih = (const float*)d_in[5];
  const float* gen_b_hh = (const float*)d_in[6];
  const float* con_w_ih = (const float*)d_in[7];
  const float* con_w_hh = (const float*)d_in[8];
  const float* con_b_ih = (const float*)d_in[9];
  const float* con_b_hh = (const float*)d_in[10];
  const float* co_w     = (const float*)d_in[11];
  const float* co_b     = (const float*)d_in[12];
  const float* fac_w    = (const float*)d_in[13];
  float* out = (float*)d_out;

  __bf16* wbf  = (__bf16*)d_ws;
  float*  invn = (float*)((char*)d_ws + WS_INVN_BYTES);

  prep_norms<<<1, 64, 0, stream>>>(fac_w, invn);
  prep_weights<<<256, 256, 0, stream>>>(con_w_ih, con_w_hh, gen_w_ih, gen_w_hh,
                                        co_w, fac_w, invn, wbf);

  int B = in_sizes[0] / X_LD;                 // 65536
  dim3 grid(B / 64), block(128);              // 4 waves x 16 batch rows
  decoder_cell<<<grid, block, 0, stream>>>(
      x, h0, eps, gen_b_ih, gen_b_hh, con_b_ih, con_b_hh, co_b, wbf, out);
}
